// NarrowSelfAttention_63737314673262
// MI455X (gfx1250) — matrix operand order repaired and checked by reading, not compiled, not run on hardware
//
#include <hip/hip_runtime.h>
#include <hip/hip_bf16.h>

typedef __attribute__((ext_vector_type(16))) __bf16 v16bf;
typedef __attribute__((ext_vector_type(8)))  __bf16 v8bf;
typedef __attribute__((ext_vector_type(8)))  float  v8f;

#define B_  4
#define T_  2048
#define C_  1024
#define H_  16
#define D_  64
#define BT_ (B_*T_)

// padded LDS row widths (in halfs) to break bank-conflict strides
#define ASW 40   // 80B rows: 16 lanes' 16B reads tile all 64 banks exactly
#define KSW 72   // 144B rows
#define PSW 72

static __device__ __forceinline__ unsigned short f2bf(float f) {
  __bf16 b = (__bf16)f;                // native RNE f32->bf16
  unsigned short u;
  __builtin_memcpy(&u, &b, 2);
  return u;
}

// A-operand: 16 contiguous halfs from `lo` (two 16B reads)
static __device__ __forceinline__ v16bf ld16_split(const unsigned short* lo,
                                                   const unsigned short* hi) {
  v8bf a = *(const v8bf*)lo;
  v8bf b = *(const v8bf*)(lo + 8);
  v16bf r;
#pragma unroll
  for (int i = 0; i < 8; ++i) { r[i] = a[i]; r[i + 8] = b[i]; }
  return r;
}
// B-operand: 16 contiguous halfs
static __device__ __forceinline__ v16bf ld16_contig(const unsigned short* p) {
  v8bf a = *(const v8bf*)p;
  v8bf b = *(const v8bf*)(p + 8);
  v16bf r;
#pragma unroll
  for (int i = 0; i < 8; ++i) { r[i] = a[i]; r[i + 8] = b[i]; }
  return r;
}

// ---------------------------------------------------------------------------
// Kernel 1: QKV projection.  out = x @ W + b, bf16, all stored [B,H,T,64].
// Block tile 128x128, k-step 32, 8 waves (2x4), wave tile 64x32.
// ---------------------------------------------------------------------------
__global__ __launch_bounds__(256)
void qkv_gemm_kernel(const float* __restrict__ x,
                     const float* __restrict__ wq, const float* __restrict__ bq,
                     const float* __restrict__ wk, const float* __restrict__ bk,
                     const float* __restrict__ wv, const float* __restrict__ bv,
                     unsigned short* __restrict__ qo,
                     unsigned short* __restrict__ ko,
                     unsigned short* __restrict__ vo) {
  __shared__ __align__(16) unsigned short As[128][ASW];   // [m][k]
  __shared__ __align__(16) unsigned short Bs[128][ASW];   // [n][k] (W tile transposed)

  const int mode = blockIdx.z;
  const float* W    = (mode == 0) ? wq : (mode == 1) ? wk : wv;
  const float* bias = (mode == 0) ? bq : (mode == 1) ? bk : bv;
  unsigned short* o = (mode == 0) ? qo : (mode == 1) ? ko : vo;

  const int n0 = blockIdx.x * 128;
  const int m0 = blockIdx.y * 128;
  const int tid  = threadIdx.x;
  const int lane = tid & 31;
  const int wave = tid >> 5;
  const int wM = (wave >> 2) * 64;
  const int wN = (wave & 3) * 32;
  const int hl = lane >> 4;          // half-wave select
  const int ln = lane & 15;
  const int kb0  = hl * 16;
  const int kb16 = hl * 16;

  v8f acc[4][2];
#pragma unroll
  for (int i = 0; i < 4; ++i)
#pragma unroll
    for (int j = 0; j < 2; ++j)
#pragma unroll
      for (int f = 0; f < 8; ++f) acc[i][j][f] = 0.f;

  // per-thread staging coordinates
  const int am[4] = { (tid + 0) >> 3, (tid + 256) >> 3, (tid + 512) >> 3, (tid + 768) >> 3 };
  const int ak    = (tid & 7) << 2;
  const int bk_[4] = { (tid + 0) >> 5, (tid + 256) >> 5, (tid + 512) >> 5, (tid + 768) >> 5 };
  const int bn    = (tid & 31) << 2;

  for (int k0 = 0; k0 < C_; k0 += 32) {
    // issue ALL global loads first, then convert (lets loadcnt drain in batch)
    float4 av[4], bvv[4];
#pragma unroll
    for (int i = 0; i < 4; ++i)
      av[i] = *(const float4*)&x[(size_t)(m0 + am[i]) * C_ + k0 + ak];
#pragma unroll
    for (int i = 0; i < 4; ++i)
      bvv[i] = *(const float4*)&W[(size_t)(k0 + bk_[i]) * C_ + n0 + bn];
    if (k0 + 32 < C_) {
      __builtin_prefetch(&x[(size_t)(m0 + (tid >> 1)) * C_ + k0 + 32], 0, 1);
      __builtin_prefetch(&W[(size_t)(k0 + 32 + (tid & 31)) * C_ + n0], 0, 1);
    }
#pragma unroll
    for (int i = 0; i < 4; ++i) {
      As[am[i]][ak + 0] = f2bf(av[i].x); As[am[i]][ak + 1] = f2bf(av[i].y);
      As[am[i]][ak + 2] = f2bf(av[i].z); As[am[i]][ak + 3] = f2bf(av[i].w);
    }
#pragma unroll
    for (int i = 0; i < 4; ++i) {
      Bs[bn + 0][bk_[i]] = f2bf(bvv[i].x); Bs[bn + 1][bk_[i]] = f2bf(bvv[i].y);
      Bs[bn + 2][bk_[i]] = f2bf(bvv[i].z); Bs[bn + 3][bk_[i]] = f2bf(bvv[i].w);
    }
    __syncthreads();

    v16bf bmat[2];
#pragma unroll
    for (int nj = 0; nj < 2; ++nj)
      bmat[nj] = ld16_contig(&Bs[wN + nj * 16 + ln][kb16]);
#pragma unroll
    for (int mi = 0; mi < 4; ++mi) {
      const int row = wM + mi * 16 + ln;
      v16bf amat = ld16_split(&As[row][kb0], &As[row][16 + kb0]);
#pragma unroll
      for (int nj = 0; nj < 2; ++nj)
        acc[mi][nj] = __builtin_amdgcn_wmma_f32_16x16x32_bf16(
            false, amat, false, bmat[nj], (short)0, acc[mi][nj], false, false);
    }
    __syncthreads();
  }

  // bias + store [B,H,T,64]
#pragma unroll
  for (int nj = 0; nj < 2; ++nj) {
    const int c = n0 + wN + nj * 16 + ln;
    const float bv_ = bias[c];
    const int h = c >> 6, d = c & 63;
#pragma unroll
    for (int mi = 0; mi < 4; ++mi) {
#pragma unroll
      for (int f = 0; f < 8; ++f) {
        const int r = m0 + wM + mi * 16 + f + (hl << 3);
        const int b = r >> 11, t = r & (T_ - 1);
        o[(size_t)((b * H_ + h) * T_ + t) * D_ + d] = f2bf(acc[mi][nj][f] + bv_);
      }
    }
  }
}

// ---------------------------------------------------------------------------
// Kernel 2: fused flash attention. Block = (b,h, 128 q rows); 8 waves x 16 rows.
// K,V both arrive [B,H,T,64]; V is transposed into LDS during staging.
// ---------------------------------------------------------------------------
__global__ __launch_bounds__(256)
void attn_kernel(const unsigned short* __restrict__ qg,
                 const unsigned short* __restrict__ kg,
                 const unsigned short* __restrict__ vg,
                 const int* __restrict__ mask,
                 unsigned short* __restrict__ yo) {
  __shared__ __align__(16) unsigned short Ks[64][KSW];      // [kt][d]
  __shared__ __align__(16) unsigned short Vt[64][KSW];      // [d][kt]
  __shared__ __align__(16) unsigned short Ps[8][16][PSW];   // per-wave P tile

  const int bh = blockIdx.y;
  const int b = bh >> 4;          // H_ = 16
  const int h = bh & 15;
  const int q0 = blockIdx.x * 128;
  const int tid = threadIdx.x, lane = tid & 31, wave = tid >> 5;
  const int hl = lane >> 4, ln = lane & 15;
  const int kb0 = hl * 16, kb16 = hl * 16;
  const int qw = q0 + wave * 16;

  const unsigned short* qb  = qg + (size_t)bh * T_ * D_;
  const unsigned short* kbp = kg + (size_t)bh * T_ * D_;
  const unsigned short* vb  = vg + (size_t)bh * T_ * D_;

  // Q strip as WMMA A-operands, resident in registers
  v16bf aq[2];
#pragma unroll
  for (int ks = 0; ks < 2; ++ks) {
    const unsigned short* p = &qb[(size_t)(qw + ln) * D_ + ks * 32];
    aq[ks] = ld16_split(p + kb0, p + 16 + kb0);
  }

  float m8[8], l8[8];
  v8f o[4];
#pragma unroll
  for (int f = 0; f < 8; ++f) { m8[f] = -3.0e38f; l8[f] = 0.f; }
#pragma unroll
  for (int dn = 0; dn < 4; ++dn)
#pragma unroll
    for (int f = 0; f < 8; ++f) o[dn][f] = 0.f;

  const float scale = 0.125f;     // 1/sqrt(64)
  const int sr[2] = { tid >> 3, (tid + 256) >> 3 };
  const int sc    = (tid & 7) << 3;

  for (int kt0 = 0; kt0 < T_; kt0 += 64) {
    // batch global loads of K chunk and V chunk
    uint4 kv[2], vv[2];
#pragma unroll
    for (int i = 0; i < 2; ++i) {
      kv[i] = *(const uint4*)&kbp[(size_t)(kt0 + sr[i]) * D_ + sc];
      vv[i] = *(const uint4*)&vb[(size_t)(kt0 + sr[i]) * D_ + sc];
    }
#pragma unroll
    for (int i = 0; i < 2; ++i) {
      *(uint4*)&Ks[sr[i]][sc] = kv[i];                  // [kt][d]
      const unsigned short* pv = (const unsigned short*)&vv[i];
#pragma unroll
      for (int j = 0; j < 8; ++j) Vt[sc + j][sr[i]] = pv[j];   // transpose -> [d][kt]
    }
    __syncthreads();

    // S = Q K^T  (4 N-subtiles x 2 k-steps)
    v8f s_acc[4];
#pragma unroll
    for (int nt = 0; nt < 4; ++nt) {
#pragma unroll
      for (int f = 0; f < 8; ++f) s_acc[nt][f] = 0.f;
#pragma unroll
      for (int ks = 0; ks < 2; ++ks) {
        v16bf bk2 = ld16_contig(&Ks[nt * 16 + ln][ks * 32 + kb16]);
        s_acc[nt] = __builtin_amdgcn_wmma_f32_16x16x32_bf16(
            false, aq[ks], false, bk2, (short)0, s_acc[nt], false, false);
      }
    }

    // scale + mask (C/D layout: row = f + 8*hl, col = ln)
#pragma unroll
    for (int nt = 0; nt < 4; ++nt) {
      const int ktgl = kt0 + nt * 16 + ln;
#pragma unroll
      for (int f = 0; f < 8; ++f) {
        const int qgl = qw + f + (hl << 3);
        const int mv = mask[(size_t)(b * T_ + qgl) * T_ + ktgl];
        s_acc[nt][f] = (mv == 0) ? -1.0e6f : s_acc[nt][f] * scale;
      }
    }

    // online softmax per row
    float alpha[8];
#pragma unroll
    for (int f = 0; f < 8; ++f) {
      float vmax = s_acc[0][f];
#pragma unroll
      for (int nt = 1; nt < 4; ++nt) vmax = fmaxf(vmax, s_acc[nt][f]);
#pragma unroll
      for (int off = 8; off >= 1; off >>= 1)
        vmax = fmaxf(vmax, __shfl_xor(vmax, off, 32));
      const float mnew = fmaxf(m8[f], vmax);
      alpha[f] = __expf(m8[f] - mnew);
      float lsum = 0.f;
#pragma unroll
      for (int nt = 0; nt < 4; ++nt) {
        const float p = __expf(s_acc[nt][f] - mnew);
        s_acc[nt][f] = p;
        lsum += p;
      }
      l8[f] = l8[f] * alpha[f] + lsum;
      m8[f] = mnew;
    }

    // P -> per-wave LDS (bf16), rescale O accumulators
#pragma unroll
    for (int nt = 0; nt < 4; ++nt)
#pragma unroll
      for (int f = 0; f < 8; ++f)
        Ps[wave][f + (hl << 3)][nt * 16 + ln] = f2bf(s_acc[nt][f]);
#pragma unroll
    for (int dn = 0; dn < 4; ++dn)
#pragma unroll
      for (int f = 0; f < 8; ++f)
        o[dn][f] *= alpha[f];

    // cross-lane LDS RAW within the wave: drain DS counter
    asm volatile("s_wait_dscnt 0" ::: "memory");

    // O += P V  (A = P from LDS, B = V^T tile [d][kt])
    v16bf ap[2];
#pragma unroll
    for (int ks = 0; ks < 2; ++ks)
      ap[ks] = ld16_split(&Ps[wave][ln][ks * 32 + kb0],
                          &Ps[wave][ln][ks * 32 + 16 + kb0]);
#pragma unroll
    for (int dn = 0; dn < 4; ++dn) {
#pragma unroll
      for (int ks = 0; ks < 2; ++ks) {
        v16bf bv2 = ld16_contig(&Vt[dn * 16 + ln][ks * 32 + kb16]);
        o[dn] = __builtin_amdgcn_wmma_f32_16x16x32_bf16(
            false, ap[ks], false, bv2, (short)0, o[dn], false, false);
      }
    }
    __syncthreads();
  }

  // normalize rows and store y as bf16 [B,T,C]
#pragma unroll
  for (int f = 0; f < 8; ++f) {
    float l = l8[f];
#pragma unroll
    for (int off = 8; off >= 1; off >>= 1) l += __shfl_xor(l, off, 32);
    l8[f] = 1.0f / l;
  }
#pragma unroll
  for (int dn = 0; dn < 4; ++dn) {
#pragma unroll
    for (int f = 0; f < 8; ++f) {
      const int qgl = qw + f + (hl << 3);
      const int c = h * D_ + dn * 16 + ln;
      yo[(size_t)(b * T_ + qgl) * C_ + c] = f2bf(o[dn][f] * l8[f]);
    }
  }
}

// ---------------------------------------------------------------------------
// Kernel 3: output projection.  out(f32) = y(bf16) @ w_o + b_o
// ---------------------------------------------------------------------------
__global__ __launch_bounds__(256)
void out_gemm_kernel(const unsigned short* __restrict__ y,
                     const float* __restrict__ wo, const float* __restrict__ bo,
                     float* __restrict__ out) {
  __shared__ __align__(16) unsigned short As[128][ASW];
  __shared__ __align__(16) unsigned short Bs[128][ASW];

  const int n0 = blockIdx.x * 128;
  const int m0 = blockIdx.y * 128;
  const int tid  = threadIdx.x;
  const int lane = tid & 31;
  const int wave = tid >> 5;
  const int wM = (wave >> 2) * 64;
  const int wN = (wave & 3) * 32;
  const int hl = lane >> 4;
  const int ln = lane & 15;
  const int kb0  = hl * 16;
  const int kb16 = hl * 16;

  v8f acc[4][2];
#pragma unroll
  for (int i = 0; i < 4; ++i)
#pragma unroll
    for (int j = 0; j < 2; ++j)
#pragma unroll
      for (int f = 0; f < 8; ++f) acc[i][j][f] = 0.f;

  const int am[2] = { (tid + 0) >> 2, (tid + 256) >> 2 };
  const int ak    = (tid & 3) << 3;
  const int bk_[4] = { (tid + 0) >> 5, (tid + 256) >> 5, (tid + 512) >> 5, (tid + 768) >> 5 };
  const int bn    = (tid & 31) << 2;

  for (int k0 = 0; k0 < C_; k0 += 32) {
    uint4 av[2];
    float4 bvv[4];
#pragma unroll
    for (int i = 0; i < 2; ++i)
      av[i] = *(const uint4*)&y[(size_t)(m0 + am[i]) * C_ + k0 + ak];
#pragma unroll
    for (int i = 0; i < 4; ++i)
      bvv[i] = *(const float4*)&wo[(size_t)(k0 + bk_[i]) * C_ + n0 + bn];
    if (k0 + 32 < C_) {
      __builtin_prefetch(&y[(size_t)(m0 + (tid >> 1)) * C_ + k0 + 32], 0, 1);
      __builtin_prefetch(&wo[(size_t)(k0 + 32 + (tid & 31)) * C_ + n0], 0, 1);
    }
#pragma unroll
    for (int i = 0; i < 2; ++i)
      *(uint4*)&As[am[i]][ak] = av[i];
#pragma unroll
    for (int i = 0; i < 4; ++i) {
      Bs[bn + 0][bk_[i]] = f2bf(bvv[i].x); Bs[bn + 1][bk_[i]] = f2bf(bvv[i].y);
      Bs[bn + 2][bk_[i]] = f2bf(bvv[i].z); Bs[bn + 3][bk_[i]] = f2bf(bvv[i].w);
    }
    __syncthreads();

    v16bf bmat[2];
#pragma unroll
    for (int nj = 0; nj < 2; ++nj)
      bmat[nj] = ld16_contig(&Bs[wN + nj * 16 + ln][kb16]);
#pragma unroll
    for (int mi = 0; mi < 4; ++mi) {
      const int row = wM + mi * 16 + ln;
      v16bf amat = ld16_split(&As[row][kb0], &As[row][16 + kb0]);
#pragma unroll
      for (int nj = 0; nj < 2; ++nj)
        acc[mi][nj] = __builtin_amdgcn_wmma_f32_16x16x32_bf16(
            false, amat, false, bmat[nj], (short)0, acc[mi][nj], false, false);
    }
    __syncthreads();
  }

#pragma unroll
  for (int nj = 0; nj < 2; ++nj) {
    const int c = n0 + wN + nj * 16 + ln;
    const float bv_ = bo[c];
#pragma unroll
    for (int mi = 0; mi < 4; ++mi) {
#pragma unroll
      for (int f = 0; f < 8; ++f) {
        const int r = m0 + wM + mi * 16 + f + (hl << 3);
        out[(size_t)r * C_ + c] = acc[mi][nj][f] + bv_;
      }
    }
  }
}

extern "C" void kernel_launch(void* const* d_in, const int* in_sizes, int n_in,
                              void* d_out, int out_size, void* d_ws, size_t ws_size,
                              hipStream_t stream) {
  const float* x  = (const float*)d_in[0];
  const int* mask = (const int*)d_in[1];
  const float* wq = (const float*)d_in[2];
  const float* bq = (const float*)d_in[3];
  const float* wk = (const float*)d_in[4];
  const float* bk = (const float*)d_in[5];
  const float* wv = (const float*)d_in[6];
  const float* bv = (const float*)d_in[7];
  const float* wo = (const float*)d_in[8];
  const float* bo = (const float*)d_in[9];
  float* out = (float*)d_out;

  const size_t elems = (size_t)B_ * H_ * T_ * D_;   // 8,388,608 per tensor
  unsigned short* qws = (unsigned short*)d_ws;
  unsigned short* kws = qws + elems;
  unsigned short* vws = kws + elems;   // [B,H,T,64] like Q/K
  unsigned short* yws = vws + elems;   // [B,T,C]

  dim3 blk(256);
  qkv_gemm_kernel<<<dim3(C_ / 128, BT_ / 128, 3), blk, 0, stream>>>(
      x, wq, bq, wk, bk, wv, bv, qws, kws, vws);
  attn_kernel<<<dim3(T_ / 128, B_ * H_), blk, 0, stream>>>(
      qws, kws, vws, mask, yws);
  out_gemm_kernel<<<dim3(C_ / 128, BT_ / 128), blk, 0, stream>>>(
      yws, wo, bo, out);
}
